// FullAttentionHead_81956565942658
// MI455X (gfx1250) — matrix operand order repaired and checked by reading, not compiled, not run on hardware
//
#include <hip/hip_runtime.h>

// MI455X / gfx1250, wave32, WMMA bf16 path.
// Fused single-head attention: qkv projection (bf16 WMMA) + flash attention.
// Scheduling is pinned with __builtin_amdgcn_sched_barrier so all WMMA
// operand loads for a K-step issue as one clause (single wait) and the
// matrix ops run back-to-back, instead of load->wait->wmma chains.

typedef __bf16 bf16;
typedef __attribute__((ext_vector_type(16))) __bf16 v16bf;
typedef __attribute__((ext_vector_type(8)))  __bf16 v8bf;
typedef __attribute__((ext_vector_type(4)))  __bf16 v4bf;
typedef __attribute__((ext_vector_type(8)))  float  v8f;
typedef __attribute__((ext_vector_type(4)))  float  v4f;

constexpr int BDIM = 8;
constexpr int SDIM = 2048;
constexpr int EDIM = 4096;
constexpr int HDIM = 128;

#if __has_builtin(__builtin_amdgcn_sched_barrier)
#define SCHED_FENCE() __builtin_amdgcn_sched_barrier(0)
#else
#define SCHED_FENCE()
#endif

__device__ __forceinline__ bf16 f2bf(float f) {
  // round-to-nearest-even fp32 -> bf16
  unsigned u = __builtin_bit_cast(unsigned, f);
  unsigned short h = (unsigned short)((u + 0x7FFFu + ((u >> 16) & 1u)) >> 16);
  return __builtin_bit_cast(bf16, h);
}

__device__ __forceinline__ v16bf cat16(v8bf lo, v8bf hi) {
  return __builtin_shufflevector(lo, hi, 0,1,2,3,4,5,6,7,8,9,10,11,12,13,14,15);
}

__device__ __forceinline__ v8bf ldg8(const bf16* p) {
  return *reinterpret_cast<const v8bf*>(p);
}

__device__ __forceinline__ v8f wmma_bf16(v16bf a, v16bf b, v8f c) {
  return __builtin_amdgcn_wmma_f32_16x16x32_bf16(
      false, a, false, b, (short)0, c, false, false);
}

// ---------------------------------------------------------------------------
// Kernel 1: convert + transpose weights: W[E][H] fp32 -> Wt[3][H][E] bf16.
// Transposed so WMMA B-fragments read two runs of 8 contiguous K values per lane.
// ---------------------------------------------------------------------------
__global__ __launch_bounds__(256)
void wconv(const float* __restrict__ wq, const float* __restrict__ wk,
           const float* __restrict__ wv, bf16* __restrict__ wt) {
  int id = blockIdx.x * 256 + threadIdx.x;          // 0 .. 3*E*H-1 exact
  int o  = id / (EDIM * HDIM);
  int r  = id % (EDIM * HDIM);
  int e  = r / HDIM;
  int h  = r % HDIM;
  const float* w = (o == 0) ? wq : ((o == 1) ? wk : wv);
  wt[((size_t)o * HDIM + h) * EDIM + e] = f2bf(w[(size_t)e * HDIM + h]);
}

// ---------------------------------------------------------------------------
// Kernel 2: fused QKV projection.  M = B*S = 16384, N = H = 128, K = E = 4096.
// Block: 256 threads = 8 waves; tile 64 rows x 128 cols x 3 outputs.
// Wave (g = wave&3, half = wave>>2): rows [16g,16g+16), cols [64*half, +64).
// x tile staged in LDS as bf16 once per K-step and reused by all three GEMMs
// (reads the 256 MiB x exactly once).  q pre-scaled by H^-1/2; v stored
// transposed [B][H][S] for contiguous PV B-fragments later.
// ---------------------------------------------------------------------------
constexpr int BM = 64;
constexpr int KC = 32;

__global__ __launch_bounds__(256, 1)
void qkv_proj(const float* __restrict__ x, const bf16* __restrict__ wt,
              bf16* __restrict__ qb, bf16* __restrict__ kb,
              bf16* __restrict__ vt) {
  __shared__ __align__(16) bf16 xs[BM * KC];
  const int tid   = threadIdx.x;
  const int lane  = tid & 31;
  const int wave  = tid >> 5;
  const int mgrp  = wave & 3;
  const int nhalf = wave >> 2;
  const int mlane = lane & 15;
  const int koffA = (lane < 16) ? 0 : 8;    // A-frag K sub-offset (16-bit layout)
  const int koffB = (lane < 16) ? 0 : 8;    // B-frag K sub-offset (16-bit layout)
  const float* xrow = x + (size_t)blockIdx.x * BM * EDIM;

  // Per-lane W row base (constant h per lane; k advances; tile offsets fold
  // into immediate offsets of the loads).
  const bf16* wb0 =
      wt + ((size_t)(nhalf * 4) * 16 + mlane) * EDIM + koffB;

  v8f acc[3][4];
#pragma unroll
  for (int o = 0; o < 3; ++o)
#pragma unroll
    for (int t = 0; t < 4; ++t) acc[o][t] = {};

  for (int k0 = 0; k0 < EDIM; k0 += KC) {
    __syncthreads();
    // stage 64x32 fp32 -> bf16: 512 float4 over 256 threads
#pragma unroll
    for (int it = 0; it < 2; ++it) {
      int idx = tid + it * 256;
      int row = idx >> 3, c4 = idx & 7;
      v4f f = *reinterpret_cast<const v4f*>(xrow + (size_t)row * EDIM + k0 + c4 * 4);
      v4bf bv;
      bv[0] = f2bf(f[0]); bv[1] = f2bf(f[1]);
      bv[2] = f2bf(f[2]); bv[3] = f2bf(f[3]);
      *reinterpret_cast<v4bf*>(&xs[row * KC + c4 * 4]) = bv;
      if (k0 + KC < EDIM)
        __builtin_prefetch(xrow + (size_t)row * EDIM + k0 + KC + c4 * 4, 0, 0);
    }
    __syncthreads();

    // ---- Operand fetch phase: A frag from LDS + all 12 B frags (24 b128). --
    const int mrow = mgrp * 16 + mlane;
    v16bf a = cat16(ldg8(&xs[mrow * KC + koffA]),
                    ldg8(&xs[mrow * KC + 16 + koffA]));
    v16bf bfrag[12];
#pragma unroll
    for (int o = 0; o < 3; ++o)
#pragma unroll
      for (int t = 0; t < 4; ++t) {
        const bf16* wrow = wb0 + ((size_t)o * HDIM + t * 16) * EDIM + k0;
        bfrag[o * 4 + t] = cat16(ldg8(wrow), ldg8(wrow + 16));
      }
    SCHED_FENCE();   // pin: all loads issue before any matrix op

    // ---- Compute phase: 12 back-to-back WMMAs, shared A operand. ----------
#pragma unroll
    for (int o = 0; o < 3; ++o)
#pragma unroll
      for (int t = 0; t < 4; ++t)
        acc[o][t] = wmma_bf16(a, bfrag[o * 4 + t], acc[o][t]);
    SCHED_FENCE();
  }

  // Epilogue.  C-tile layout: VGPR r, lanes 0-15 -> row r, lanes 16-31 -> row r+8.
  const int rowadd = (lane < 16) ? 0 : 8;
  const size_t gm = (size_t)blockIdx.x * BM + mgrp * 16;   // flat row over B*S
  const int bidx = (int)((blockIdx.x * BM) / SDIM);
  const int s0   = (int)((blockIdx.x * BM) % SDIM) + mgrp * 16 + rowadd;
  const float qscale = 0.08838834764831845f;               // H^-0.5
#pragma unroll
  for (int t = 0; t < 4; ++t) {
    int h = (nhalf * 4 + t) * 16 + mlane;
#pragma unroll
    for (int r = 0; r < 8; ++r) {
      size_t row = gm + r + rowadd;
      qb[row * HDIM + h] = f2bf(acc[0][t][r] * qscale);
      kb[row * HDIM + h] = f2bf(acc[1][t][r]);
    }
    v8bf pk;
#pragma unroll
    for (int r = 0; r < 8; ++r) pk[r] = f2bf(acc[2][t][r]);
    *reinterpret_cast<v8bf*>(vt + ((size_t)bidx * HDIM + h) * SDIM + s0) = pk;
  }
}

// ---------------------------------------------------------------------------
// Kernel 3: flash attention.  Grid (S/64, B), 128 threads = 4 waves,
// each wave owns 16 query rows.  Per 32-key block:
//   * 16 K-frag loads clustered -> 8 back-to-back WMMAs for Q@K^T
//   * 16 V-frag loads issued right after (K regs dead), latency hidden
//     under the softmax VALU/transcendental block
//   * online softmax with __shfl_xor row reductions (masks <=8 stay inside
//     the 16-lane N-group matching the C layout)
//   * P re-laid out C-tile -> A-frag through per-wave LDS
//   * 8 WMMAs for P@V (v stored transposed)
// ---------------------------------------------------------------------------
__global__ __launch_bounds__(128, 1)
void flash_attn(const bf16* __restrict__ qb, const bf16* __restrict__ kb,
                const bf16* __restrict__ vt, float* __restrict__ out) {
  __shared__ __align__(16) bf16 pshare[4][16 * 32];
  const int b     = blockIdx.y;
  const int lane  = threadIdx.x & 31;
  const int wave  = threadIdx.x >> 5;
  const int q0    = blockIdx.x * 64 + wave * 16;
  const int mlane = lane & 15;
  const int koffA = (lane < 16) ? 0 : 8;
  const int koffB = (lane < 16) ? 0 : 8;
  const int rowadd = (lane < 16) ? 0 : 8;
  bf16* pbuf = pshare[wave];

  // Resident Q A-fragments: 16 rows x H=128 -> 4 frags of 16x32.
  const bf16* qbase = qb + ((size_t)b * SDIM + q0) * HDIM;
  v16bf qa[4];
#pragma unroll
  for (int f = 0; f < 4; ++f) {
    const bf16* p = qbase + (size_t)mlane * HDIM + f * 32 + koffA;
    qa[f] = cat16(ldg8(p), ldg8(p + 16));
  }

  v8f oacc[8];
#pragma unroll
  for (int t = 0; t < 8; ++t) oacc[t] = {};
  float m[8], l[8];
#pragma unroll
  for (int r = 0; r < 8; ++r) { m[r] = -3.0e38f; l[r] = 0.f; }

  const bf16* kbase0 = kb + (size_t)b * SDIM * HDIM + (size_t)mlane * HDIM + koffB;
  const bf16* vbase0 = vt + (size_t)b * HDIM * SDIM + (size_t)mlane * SDIM + koffB;

  for (int j0 = 0; j0 < SDIM; j0 += 32) {
    // ---- K-frag fetch phase (one clause, one wait). ----
    const bf16* kbase = kbase0 + (size_t)j0 * HDIM;
    v16bf kfrag[8];
#pragma unroll
    for (int f = 0; f < 4; ++f) {
      const bf16* kr0 = kbase + f * 32;
      const bf16* kr1 = kr0 + 16 * HDIM;
      kfrag[f]     = cat16(ldg8(kr0), ldg8(kr0 + 16));
      kfrag[f + 4] = cat16(ldg8(kr1), ldg8(kr1 + 16));
    }
    SCHED_FENCE();

    // ---- Q@K^T: 8 back-to-back WMMAs (q pre-scaled by H^-1/2). ----
    v8f sc0 = {}, sc1 = {};
#pragma unroll
    for (int f = 0; f < 4; ++f) {
      sc0 = wmma_bf16(qa[f], kfrag[f], sc0);
      sc1 = wmma_bf16(qa[f], kfrag[f + 4], sc1);
    }
    SCHED_FENCE();   // K-frag registers die here, before V-frags go live

    // ---- V-frag fetch phase; softmax below hides the latency. ----
    v16bf vfrag[8];
#pragma unroll
    for (int t = 0; t < 8; ++t) {
      const bf16* vr = vbase0 + (size_t)(t * 16) * SDIM + j0;
      vfrag[t] = cat16(ldg8(vr), ldg8(vr + 16));
    }
    SCHED_FENCE();   // pin V loads ahead of the softmax VALU block

    // ---- Online softmax: per-row max/sum across the 16-lane N group. ----
    float corr[8];
#pragma unroll
    for (int r = 0; r < 8; ++r) {
      float smax = fmaxf(sc0[r], sc1[r]);
      smax = fmaxf(smax, __shfl_xor(smax, 1));
      smax = fmaxf(smax, __shfl_xor(smax, 2));
      smax = fmaxf(smax, __shfl_xor(smax, 4));
      smax = fmaxf(smax, __shfl_xor(smax, 8));
      float mnew = fmaxf(m[r], smax);
      float c  = __expf(m[r] - mnew);
      float p0 = __expf(sc0[r] - mnew);
      float p1 = __expf(sc1[r] - mnew);
      sc0[r] = p0; sc1[r] = p1;
      float rs = p0 + p1;
      rs += __shfl_xor(rs, 1);
      rs += __shfl_xor(rs, 2);
      rs += __shfl_xor(rs, 4);
      rs += __shfl_xor(rs, 8);
      l[r] = l[r] * c + rs;
      m[r] = mnew;
      corr[r] = c;
    }
#pragma unroll
    for (int t = 0; t < 8; ++t)
#pragma unroll
      for (int r = 0; r < 8; ++r) oacc[t][r] *= corr[r];

    // ---- P: C-tile layout -> A-fragment layout via per-wave LDS bounce. ----
#pragma unroll
    for (int r = 0; r < 8; ++r) {
      pbuf[(r + rowadd) * 32 + mlane]      = f2bf(sc0[r]);
      pbuf[(r + rowadd) * 32 + 16 + mlane] = f2bf(sc1[r]);
    }
    __asm__ volatile("s_wait_dscnt 0" ::: "memory");
    v16bf pa = cat16(ldg8(&pbuf[mlane * 32 + koffA]),
                     ldg8(&pbuf[mlane * 32 + 16 + koffA]));

    // ---- out += P @ V : operands already resident. ----
#pragma unroll
    for (int t = 0; t < 8; ++t)
      oacc[t] = wmma_bf16(pa, vfrag[t], oacc[t]);
    SCHED_FENCE();
  }

  float* obase = out + ((size_t)b * SDIM + q0) * HDIM;
#pragma unroll
  for (int r = 0; r < 8; ++r) {
    float inv = 1.f / l[r];
#pragma unroll
    for (int t = 0; t < 8; ++t)
      obase[(size_t)(r + rowadd) * HDIM + t * 16 + mlane] = oacc[t][r] * inv;
  }
}

// ---------------------------------------------------------------------------
extern "C" void kernel_launch(void* const* d_in, const int* in_sizes, int n_in,
                              void* d_out, int out_size, void* d_ws, size_t ws_size,
                              hipStream_t stream) {
  (void)in_sizes; (void)n_in; (void)out_size; (void)ws_size;
  const float* x  = (const float*)d_in[0];
  const float* wq = (const float*)d_in[1];
  const float* wk = (const float*)d_in[2];
  const float* wv = (const float*)d_in[3];
  float* out = (float*)d_out;

  // Workspace layout (bf16): Wt[3][H][E] | q[B*S][H] | k[B*S][H] | vT[B][H][S]
  bf16* wt   = reinterpret_cast<bf16*>(d_ws);
  bf16* qbuf = wt   + (size_t)3 * HDIM * EDIM;
  bf16* kbuf = qbuf + (size_t)BDIM * SDIM * HDIM;
  bf16* vbuf = kbuf + (size_t)BDIM * SDIM * HDIM;

  wconv<<<(3 * EDIM * HDIM) / 256, 256, 0, stream>>>(wq, wk, wv, wt);
  qkv_proj<<<(BDIM * SDIM) / BM, 256, 0, stream>>>(x, wt, qbuf, kbuf, vbuf);
  flash_attn<<<dim3(SDIM / 64, BDIM), 128, 0, stream>>>(qbuf, kbuf, vbuf, out);
}
